// TextEncoder_17403207483648
// MI455X (gfx1250) — matrix operand order repaired and checked by reading, not compiled, not run on hardware
//
#include <hip/hip_runtime.h>
#include <hip/hip_bf16.h>
#include <stdint.h>

// ---------------------------------------------------------------------------
// B=16, N=20, D=300, H=256, NT=30, SEM=128. 4H=1024. Spans/batch=190.
// Rows = 3040. Out cols = 3840.
// ---------------------------------------------------------------------------

typedef __attribute__((ext_vector_type(16))) __bf16 v16bf;
typedef __attribute__((ext_vector_type(8)))  float  v8f;
typedef __attribute__((ext_vector_type(4)))  float  fv4;
typedef __attribute__((ext_vector_type(4)))  int    iv4;

union FragBF {            // 32 bytes = 8 VGPRs worth of a 16x32 bf16 fragment
    iv4   q[2];
    v16bf v;
};

__device__ __forceinline__ unsigned short f2bf(float f) {
    unsigned int u = __float_as_uint(f);
    unsigned int r = u + 0x7FFFu + ((u >> 16) & 1u);   // round-to-nearest-even
    return (unsigned short)(r >> 16);
}

#define LOG2E 1.4426950408889634f
// branch-free activations on v_exp_f32 / v_rcp_f32 (no EXEC divergence)
__device__ __forceinline__ float fsigmoid(float x) {
    return __builtin_amdgcn_rcpf(1.0f + __builtin_amdgcn_exp2f(-LOG2E * x));
}
__device__ __forceinline__ float ftanh(float x) {
    return 2.0f * __builtin_amdgcn_rcpf(1.0f + __builtin_amdgcn_exp2f(-2.0f * LOG2E * x)) - 1.0f;
}

// ---------------------------------------------------------------------------
// Phase 0: xg contributions, gate-interleaved layout [dir][b][t][hcol(256)][g(4)]
// grid = 640 blocks x 256 thr ; float4-vectorized 300-dot
// ---------------------------------------------------------------------------
__global__ __launch_bounds__(256)
void xg_kernel(const int* __restrict__ x, const int* __restrict__ lengths,
               const float* __restrict__ emb,
               const float* __restrict__ Wif, const float* __restrict__ bfv,
               const float* __restrict__ Wib, const float* __restrict__ bbv,
               float* __restrict__ xg) {
    int bt  = blockIdx.x;          // dir*320 + b*20 + t
    int dir = bt / 320;
    int rem = bt % 320;
    int b   = rem / 20;
    int t   = rem % 20;

    __shared__ __align__(16) float e[300];
    fv4* eq = (fv4*)e;
    int  tid    = threadIdx.x;
    bool masked = (t >= lengths[b]);
    int  tok    = x[b * 20 + t];
    const fv4* embq = (const fv4*)emb;              // 75 fv4 per row (1200 B, aligned)
    for (int i = tid; i < 75; i += 256) {
        fv4 v = embq[(size_t)tok * 75 + i];
        if (masked) { v[0] = 0.f; v[1] = 0.f; v[2] = 0.f; v[3] = 0.f; }
        eq[i] = v;
    }
    __syncthreads();

    const float* Wi   = dir ? Wib : Wif;
    const float* bias = dir ? bbv : bfv;
    const fv4*   Wq   = (const fv4*)Wi;
    for (int c = tid; c < 1024; c += 256) {         // c = gate row in 4H
        float acc = bias[c];
        const fv4* wr = Wq + (size_t)c * 75;
        for (int d = 0; d < 75; ++d) {
            fv4 w = wr[d], ev = eq[d];
            acc += w[0]*ev[0] + w[1]*ev[1] + w[2]*ev[2] + w[3]*ev[3];
        }
        int g   = c >> 8;
        int col = c & 255;
        xg[((size_t)bt * 256 + col) * 4 + g] = acc;
    }
}

// ---------------------------------------------------------------------------
// Pack W_hh (2 x [1024 x 256]) into bf16 WMMA B-fragment tiles (32K x 16N).
// Layout: [dir(2)][nt(64)][kt(8)][lane(32)][v(8)] dwords.   total = 262144
// ---------------------------------------------------------------------------
__global__ __launch_bounds__(256)
void pack_wh_kernel(const float* __restrict__ Whf, const float* __restrict__ Whb,
                    unsigned int* __restrict__ packed) {
    int idx = blockIdx.x * 256 + threadIdx.x;      // 18 bits
    int v    = idx & 7;
    int lane = (idx >> 3) & 31;
    int kt   = (idx >> 8) & 7;
    int nt   = (idx >> 11) & 63;
    int dir  = idx >> 17;
    const float* W = dir ? Whb : Whf;              // (1024 x 256) row-major
    int n = nt * 16 + (lane & 15);
    int k = kt * 32 + (v >> 2) * 16 + (lane >> 4) * 8 + 2 * (v & 3);
    unsigned int lo = f2bf(W[n * 256 + k]);
    unsigned int hi = f2bf(W[n * 256 + k + 1]);
    packed[idx] = lo | (hi << 16);
}

// ---------------------------------------------------------------------------
// Pack W_out (3840 x 512) into bf16 B-fragment tiles.
// Layout: [nt(240)][kt(16)][lane(32)][v(8)] dwords.  total = 983040
// ---------------------------------------------------------------------------
__global__ __launch_bounds__(256)
void pack_wo_kernel(const float* __restrict__ Wout, unsigned int* __restrict__ packed) {
    int idx = blockIdx.x * 256 + threadIdx.x;
    int v    = idx & 7;
    int lane = (idx >> 3) & 31;
    int kt   = (idx >> 8) & 15;
    int nt   = idx >> 12;                          // < 240
    int n = nt * 16 + (lane & 15);
    int k = kt * 32 + (v >> 2) * 16 + (lane >> 4) * 8 + 2 * (v & 3);
    unsigned int lo = f2bf(Wout[n * 512 + k]);
    unsigned int hi = f2bf(Wout[n * 512 + k + 1]);
    packed[idx] = lo | (hi << 16);
}

// ---------------------------------------------------------------------------
// Phase 1: batched BiLSTM recurrence, TWO span-tiles per block.
// Block = (dir, k, {s0, s0+1}); each tile = 16 spans (rows = batch).
// Wave w owns hidden cols [16w,16w+16).  Each B-fragment load from L2 feeds
// two WMMAs (tile0 + tile1) -> halves weight traffic.  h double-buffered in
// LDS (bf16); kt loop kept rolled so weights are not hoisted/spilled.
// If (20-k) is odd the last block processes its single tile twice
// (idempotent duplicate stores -> deterministic).
// grid = 200 blocks x 512 thr
// ---------------------------------------------------------------------------
__global__ __launch_bounds__(512)
void lstm_kernel(const float* __restrict__ xg,
                 const unsigned int* __restrict__ packedWh,
                 unsigned short* __restrict__ hsum) {
    int blk = blockIdx.x;             // 0..199
    int dir = blk / 100;
    int j   = blk % 100;
    int k = 1, off = 0;
    for (;;) {
        int nb = (21 - k) >> 1;       // ceil((20-k)/2) tile-pairs for this k
        if (j < nb) break;
        j -= nb; off += 20 - k; ++k;
    }
    int ns = 20 - k;                  // starts for this k
    int s0 = 2 * j;
    int s1 = (s0 + 1 < ns) ? s0 + 1 : s0;
    int p0 = off + s0, p1 = off + s1; // span indices within batch
    int L = k + 1;

    int wave = threadIdx.x >> 5;
    int lane = threadIdx.x & 31;
    int half = lane >> 4;
    int n    = lane & 15;

    // [buf(2)][tile(2)][16 rows][256 cols] bf16 = 32 KB
    __shared__ unsigned short h_lds[2 * 2 * 16 * 256];
    {   // zero buffer 0 (both tiles), read at t=0
        unsigned int* z = (unsigned int*)h_lds;
        for (int i = threadIdx.x; i < 4096; i += 512) z[i] = 0u;
    }
    __syncthreads();

    float c_st[2][8], hs[2][8];
#pragma unroll
    for (int tau = 0; tau < 2; ++tau)
#pragma unroll
        for (int e = 0; e < 8; ++e) { c_st[tau][e] = 0.0f; hs[tau][e] = 0.0f; }

    const fv4* xq = ((const fv4*)xg) + (size_t)dir * 320 * 256;
    const iv4* Wq = (const iv4*)packedWh;
    const iv4* Hq = (const iv4*)h_lds;

    int wbase = (dir * 64 + wave) * 8;   // B-tile index base for gate 0

    for (int t = 0; t < L; ++t) {
        int tok0 = dir ? (s0 + k - t) : (s0 + t);
        int tok1 = dir ? (s1 + k - t) : (s1 + t);
        int rb   = t & 1;                // read buffer
        int wb   = rb ^ 1;               // write buffer

        // init gate accumulators with input contributions (D-layout, float4)
        v8f acc[2][4];
#pragma unroll
        for (int e = 0; e < 8; ++e) {
            int m = e + 8 * half;
            fv4 x0 = xq[(size_t)(m * 20 + tok0) * 256 + wave * 16 + n];
            fv4 x1 = xq[(size_t)(m * 20 + tok1) * 256 + wave * 16 + n];
#pragma unroll
            for (int g = 0; g < 4; ++g) { acc[0][g][e] = x0[g]; acc[1][g][e] = x1[g]; }
        }

        // gates += h @ Whᵀ : 8 kt iters; each B frag feeds both tiles
#pragma unroll 1
        for (int kt = 0; kt < 8; ++kt) {
            FragBF a0, a1;
            int ab = rb * 1024 + n * 32 + kt * 4 + half;   // iv4 units
            a0.q[0] = Hq[ab];        a0.q[1] = Hq[ab + 2];
            a1.q[0] = Hq[ab + 512];  a1.q[1] = Hq[ab + 514];
#pragma unroll
            for (int g = 0; g < 4; ++g) {
                FragBF bfr;
                int tIdx = wbase + g * 128 + kt;   // (dir*64 + g*16+wave)*8 + kt
                bfr.q[0] = Wq[tIdx * 64 + lane * 2];
                bfr.q[1] = Wq[tIdx * 64 + lane * 2 + 1];
                acc[0][g] = __builtin_amdgcn_wmma_f32_16x16x32_bf16(
                    false, a0.v, false, bfr.v, (short)0, acc[0][g], false, false);
                acc[1][g] = __builtin_amdgcn_wmma_f32_16x16x32_bf16(
                    false, a1.v, false, bfr.v, (short)0, acc[1][g], false, false);
            }
        }

        // elementwise LSTM cell (branch-free activations); write h to buffer wb
#pragma unroll
        for (int tau = 0; tau < 2; ++tau) {
#pragma unroll
            for (int e = 0; e < 8; ++e) {
                float iv = fsigmoid(acc[tau][0][e]);
                float fv = fsigmoid(acc[tau][1][e]);
                float gv = ftanh(acc[tau][2][e]);
                float ov = fsigmoid(acc[tau][3][e]);
                float c  = fv * c_st[tau][e] + iv * gv;
                c_st[tau][e] = c;
                float h  = ov * ftanh(c);
                hs[tau][e] += h;
                int m = e + 8 * half;
                h_lds[(wb * 2 + tau) * 4096 + m * 256 + wave * 16 + n] = f2bf(h);
            }
        }
        __syncthreads();   // covers RAW (next reads of wb) and WAR (rb reuse)
    }

    // hsum row = b*190 + p ; cols [dir*256 + wave*16 + n]
#pragma unroll
    for (int tau = 0; tau < 2; ++tau) {
        int p = tau ? p1 : p0;
#pragma unroll
        for (int e = 0; e < 8; ++e) {
            int m   = e + 8 * half;                // batch index
            int row = m * 190 + p;
            hsum[(size_t)row * 512 + dir * 256 + wave * 16 + n] = f2bf(hs[tau][e]);
        }
    }
}

// ---------------------------------------------------------------------------
// Phase 2: out_raw = hsum(3040x512) @ W_outᵀ + L·b_out   (bf16 WMMA, f32 acc)
// 190 M-tiles x 240 N-tiles = 45600 tiles; 1 wave/tile, 4 waves/block.
// grid = 11400 x 128
// ---------------------------------------------------------------------------
__global__ __launch_bounds__(128)
void proj_kernel(const unsigned short* __restrict__ hsum,
                 const unsigned int* __restrict__ packedWo,
                 const float* __restrict__ b_out,
                 float* __restrict__ out) {
    int wave = threadIdx.x >> 5;
    int lane = threadIdx.x & 31;
    int tile = blockIdx.x * 4 + wave;     // 0..45599
    int mt = tile / 240;
    int nt = tile % 240;
    int half = lane >> 4;
    int n    = lane & 15;

    const iv4* Hq = (const iv4*)hsum;     // 64 iv4 per 512-col row
    const iv4* Wq = (const iv4*)packedWo;

    v8f acc;
#pragma unroll
    for (int e = 0; e < 8; ++e) acc[e] = 0.0f;

    int arow = mt * 16 + n;               // A-fragment row for this lane
#pragma unroll 4
    for (int kt = 0; kt < 16; ++kt) {
        FragBF a, b;
        int abase = arow * 64 + kt * 4 + half;
        a.q[0] = Hq[abase];
        a.q[1] = Hq[abase + 2];
        int tIdx = nt * 16 + kt;
        b.q[0] = Wq[tIdx * 64 + lane * 2];
        b.q[1] = Wq[tIdx * 64 + lane * 2 + 1];
        acc = __builtin_amdgcn_wmma_f32_16x16x32_bf16(
            false, a.v, false, b.v, (short)0, acc, false, false);
    }

    int col = nt * 16 + n;
    float bo = b_out[col];
#pragma unroll
    for (int e = 0; e < 8; ++e) {
        int m   = e + 8 * half;
        int row = mt * 16 + m;            // 0..3039
        int pp  = row % 190;
        int kk = 1, o2 = 0;
        while (pp - o2 >= 20 - kk) { o2 += 20 - kk; ++kk; }   // span length-1
        out[(size_t)row * 3840 + col] = acc[e] + (float)(kk + 1) * bo;
    }
}

// ---------------------------------------------------------------------------
// Phase 3: in-place L2 normalization over SEM=128 per (row, nt).
// 3040*30 = 91200 groups; 1 wave each; 8 waves/block -> 11400 blocks exact.
// ---------------------------------------------------------------------------
__global__ __launch_bounds__(256)
void norm_kernel(float* __restrict__ out) {
    int gwave = (blockIdx.x * 256 + threadIdx.x) >> 5;
    int lane  = threadIdx.x & 31;
    float* base = out + (size_t)gwave * 128;
    float v0 = base[lane];
    float v1 = base[lane + 32];
    float v2 = base[lane + 64];
    float v3 = base[lane + 96];
    float ss = v0 * v0 + v1 * v1 + v2 * v2 + v3 * v3;
#pragma unroll
    for (int d = 16; d >= 1; d >>= 1) ss += __shfl_xor(ss, d, 32);
    float inv = rsqrtf(ss);
    base[lane]      = v0 * inv;
    base[lane + 32] = v1 * inv;
    base[lane + 64] = v2 * inv;
    base[lane + 96] = v3 * inv;
}

// ---------------------------------------------------------------------------
extern "C" void kernel_launch(void* const* d_in, const int* in_sizes, int n_in,
                              void* d_out, int out_size, void* d_ws, size_t ws_size,
                              hipStream_t stream) {
    const int*   x       = (const int*)d_in[0];
    const int*   lengths = (const int*)d_in[1];
    /* spans d_in[2] unused */
    const float* emb  = (const float*)d_in[3];
    const float* Wif  = (const float*)d_in[4];
    const float* Whf  = (const float*)d_in[5];
    const float* bfv  = (const float*)d_in[6];
    const float* Wib  = (const float*)d_in[7];
    const float* Whb  = (const float*)d_in[8];
    const float* bbv  = (const float*)d_in[9];
    const float* Wout = (const float*)d_in[10];
    const float* bout = (const float*)d_in[11];
    float* out = (float*)d_out;

    char* ws = (char*)d_ws;
    float*          xg       = (float*)(ws);                        // 2,621,440 B
    unsigned int*   packedWh = (unsigned int*)(ws + 2621440);       // 1,048,576 B
    unsigned int*   packedWo = (unsigned int*)(ws + 3670016);       // 3,932,160 B
    unsigned short* hsum     = (unsigned short*)(ws + 7602176);     // 3,112,960 B

    xg_kernel     <<<640,   256, 0, stream>>>(x, lengths, emb, Wif, bfv, Wib, bbv, xg);
    pack_wh_kernel<<<1024,  256, 0, stream>>>(Whf, Whb, packedWh);
    pack_wo_kernel<<<3840,  256, 0, stream>>>(Wout, packedWo);
    lstm_kernel   <<<200,   512, 0, stream>>>(xg, packedWh, hsum);
    proj_kernel   <<<11400, 128, 0, stream>>>(hsum, packedWo, bout, out);
    norm_kernel   <<<11400, 256, 0, stream>>>(out);
}
